// ExpressionAutoDiscretization_66443144069353
// MI455X (gfx1250) — matrix operand order repaired and checked by reading, not compiled, not run on hardware
//
#include <hip/hip_runtime.h>
#include <hip/hip_bf16.h>
#include <stdint.h>

typedef _Float16 f16x16 __attribute__((ext_vector_type(16)));
typedef _Float16 f16x8  __attribute__((ext_vector_type(8)));
typedef float    f32x8  __attribute__((ext_vector_type(8)));
typedef int      v4i    __attribute__((ext_vector_type(4)));

#define BINS 100
#define KPAD 128            // K padded to 4 x 32
#define NPAD 112            // 7 N-tiles of 16 for the bin dimension
#define DIM  768            // 48 N-tiles of 16
#define NTOK 154112         // 8 * 19264 tokens
#define TOKS_PER_BLOCK 128  // 8 waves x 16 tokens
#define NBLOCKS (NTOK / TOKS_PER_BLOCK)  // 1204 exactly

#if __has_builtin(__builtin_amdgcn_global_load_async_to_lds_b128)
#define EAD_ASYNC_STAGE 1
typedef __attribute__((address_space(1))) v4i* gv4i_p;   // global src
typedef __attribute__((address_space(3))) v4i* lv4i_p;   // LDS dst
#else
#define EAD_ASYNC_STAGE 0
#endif

__device__ __forceinline__ float bf16_round(float x) {
  return __bfloat162float(__float2bfloat16(x));
}

__device__ __forceinline__ f16x16 cat8(f16x8 lo, f16x8 hi) {
  return __builtin_shufflevector(lo, hi, 0,1,2,3,4,5,6,7,8,9,10,11,12,13,14,15);
}

// ---------------------------------------------------------------------------
// Prep: convert w2 (+identity, folds the alpha*v2 skip term) and the
// transposed embedding table to f16, K-padded to 128, into workspace.
// ---------------------------------------------------------------------------
__global__ void ead_prep(const float* __restrict__ w2,
                         const float* __restrict__ emb,
                         _Float16* __restrict__ w2h,
                         _Float16* __restrict__ embTh) {
  int i = blockIdx.x * 256 + threadIdx.x;
  if (i < NPAD * KPAD) {
    int j = i >> 7, k = i & 127;
    float v = 0.f;
    if (j < BINS && k < BINS) {
      v = w2[j * BINS + k];
      if (j == k) v += 1.0f;          // BIN_ALPHA = 1.0 skip connection
    }
    w2h[i] = (_Float16)v;
  }
  if (i < DIM * KPAD) {
    int d = i >> 7, k = i & 127;
    float v = (k < BINS) ? emb[(size_t)k * DIM + d] : 0.f;   // transpose
    embTh[i] = (_Float16)v;
  }
}

// ---------------------------------------------------------------------------
// Fused main kernel: v2 -> (v2 @ (w2+I)^T + b2) -> softmax -> @ emb_table
// -> pad/mask select -> f32 out.  8 waves, each owns 16 tokens.
// The 192KB embedding panel streams L2->LDS asynchronously (ASYNCcnt)
// while matmul1 + softmax run against the already-landed w2 panel.
// ---------------------------------------------------------------------------
__global__ __launch_bounds__(256)
void ead_main(const float* __restrict__ gene,
              const int*   __restrict__ padm,
              const int*   __restrict__ mskm,
              const float* __restrict__ w1,
              const float* __restrict__ b1,
              const float* __restrict__ b2,
              const float* __restrict__ pade,
              const float* __restrict__ maske,
              const _Float16* __restrict__ w2h,
              const _Float16* __restrict__ embTh,
              float* __restrict__ out) {
  __shared__ _Float16 s_emb[DIM * KPAD];    // 192 KB embedding panel (N-major)
  __shared__ _Float16 s_w2[NPAD * KPAD];    //  28 KB w2+I panel (N-major)
  __shared__ _Float16 s_A[8 * 16 * KPAD];   //  32 KB per-wave A tiles
  __shared__ float s_pad[DIM], s_mask[DIM];
  __shared__ float s_b2[NPAD], s_w1[BINS], s_b1[BINS], s_x[TOKS_PER_BLOCK];
  __shared__ int   s_fl[TOKS_PER_BLOCK];

  const int tid = threadIdx.x;
  const int tokBlock = blockIdx.x * TOKS_PER_BLOCK;

  // ---- stage panels into LDS ----
#if EAD_ASYNC_STAGE
  {
    // w2 panel first (7 x b128/thread), then embedding panel (48 x b128/thread).
    // Async loads retire in order => asynccnt<=48 proves the w2 chunks landed.
    uintptr_t gw2 = (uintptr_t)w2h + (uintptr_t)(tid * 16);
    uint32_t  lw2 = (uint32_t)(uintptr_t)(char*)s_w2 + (uint32_t)(tid * 16);
    #pragma unroll
    for (int i = 0; i < (NPAD * KPAD * 2 / 16) / 256; ++i)       // 7
      __builtin_amdgcn_global_load_async_to_lds_b128(
          (gv4i_p)(gw2 + (uintptr_t)(i * 4096)),
          (lv4i_p)(lw2 + (uint32_t)(i * 4096)), 0, 0);
    uintptr_t gem = (uintptr_t)embTh + (uintptr_t)(tid * 16);
    uint32_t  lem = (uint32_t)(uintptr_t)(char*)s_emb + (uint32_t)(tid * 16);
    #pragma unroll
    for (int i = 0; i < (DIM * KPAD * 2 / 16) / 256; ++i)        // 48
      __builtin_amdgcn_global_load_async_to_lds_b128(
          (gv4i_p)(gem + (uintptr_t)(i * 4096)),
          (lv4i_p)(lem + (uint32_t)(i * 4096)), 0, 0);
  }
#else
  {
    const uint4* src = (const uint4*)embTh;
    uint4* dst = (uint4*)s_emb;
    #pragma unroll
    for (int i = 0; i < (DIM * KPAD * 2 / 16) / 256; ++i)
      dst[tid + i * 256] = src[tid + i * 256];
    const uint4* src2 = (const uint4*)w2h;
    uint4* dst2 = (uint4*)s_w2;
    #pragma unroll
    for (int i = 0; i < (NPAD * KPAD * 2 / 16) / 256; ++i)
      dst2[tid + i * 256] = src2[tid + i * 256];
  }
#endif
  for (int i = tid; i < DIM; i += 256) {
    s_pad[i]  = bf16_round(pade[i]);     // mixed_precision bf16 round-trip
    s_mask[i] = bf16_round(maske[i]);
  }
  if (tid < NPAD) s_b2[tid] = (tid < BINS) ? b2[tid] : 0.f;
  if (tid < BINS) { s_w1[tid] = w1[tid]; s_b1[tid] = b1[tid]; }
  if (tid < TOKS_PER_BLOCK) {
    int gt = tokBlock + tid;
    s_x[tid] = gene[gt];
    s_fl[tid] = (mskm[gt] != 0 ? 2 : 0) | (padm[gt] != 0 ? 1 : 0);
  }
#if EAD_ASYNC_STAGE
  asm volatile("s_wait_asynccnt 48" ::: "memory");  // w2 panel resident
#endif
  __syncthreads();

  const int wv   = tid >> 5;
  const int lane = tid & 31;
  const int l16  = lane & 15;
  const int h    = lane >> 4;
  const int tokBase = tokBlock + wv * 16;
  _Float16* Aw = &s_A[wv * 16 * KPAD];

  // ---- v2 = LeakyReLU(x*w1 + b1), f16, zero-padded K ----
  for (int idx = lane; idx < 16 * KPAD; idx += 32) {
    int m = idx >> 7, k = idx & 127;
    float v = 0.f;
    if (k < BINS) {
      float t = s_x[wv * 16 + m] * s_w1[k] + s_b1[k];
      v = (t > 0.f) ? t : 0.1f * t;
    }
    Aw[idx] = (_Float16)v;
  }

  // ---- A fragments (16x32 f16 layout: two 16B chunks per lane) ----
  f16x16 afr[4];
  #pragma unroll
  for (int ks = 0; ks < 4; ++ks) {
    const _Float16* p = Aw + l16 * KPAD + ks * 32 + h * 8;
    afr[ks] = cat8(*(const f16x8*)p, *(const f16x8*)(p + 16));
  }

  // ---- matmul1: v3 = v2 @ (w2+I)^T + b2 over 7 N-tiles ----
  float v3v[7][8];
  #pragma unroll
  for (int jt = 0; jt < 7; ++jt) {
    f32x8 c = {};
    #pragma unroll
    for (int ks = 0; ks < 4; ++ks) {
      const _Float16* p = &s_w2[(jt * 16 + l16) * KPAD + ks * 32 + h * 8];
      f16x16 bfr = cat8(*(const f16x8*)p, *(const f16x8*)(p + 16));
      c = __builtin_amdgcn_wmma_f32_16x16x32_f16(false, afr[ks], false, bfr,
                                                 (short)0, c, false, false);
    }
    float bias = s_b2[jt * 16 + l16];
    #pragma unroll
    for (int r = 0; r < 8; ++r) v3v[jt][r] = c[r] + bias;
  }

  // ---- softmax over 100 bins (rows live in 16-lane groups) ----
  #pragma unroll
  for (int r = 0; r < 8; ++r) {
    const int m = r + 8 * h;
    float mx = -3.0e38f;
    #pragma unroll
    for (int t = 0; t < 7; ++t)
      if ((t < 6) || (l16 < 4)) mx = fmaxf(mx, v3v[t][r]);
    #pragma unroll
    for (int off = 1; off < 16; off <<= 1)
      mx = fmaxf(mx, __shfl_xor(mx, off, 16));
    float ev[7], sum = 0.f;
    #pragma unroll
    for (int t = 0; t < 7; ++t) {
      ev[t] = ((t < 6) || (l16 < 4)) ? __expf(v3v[t][r] - mx) : 0.f;
      sum += ev[t];
    }
    #pragma unroll
    for (int off = 1; off < 16; off <<= 1)
      sum += __shfl_xor(sum, off, 16);
    float inv = 1.0f / sum;
    #pragma unroll
    for (int t = 0; t < 7; ++t)
      if ((t < 6) || (l16 < 4))
        Aw[m * KPAD + t * 16 + l16] = (_Float16)(ev[t] * inv);
    // K pads (>=100) stay zero from the v2 fill.
  }

  // ---- reload A fragments (now softmax weights) ----
  #pragma unroll
  for (int ks = 0; ks < 4; ++ks) {
    const _Float16* p = Aw + l16 * KPAD + ks * 32 + h * 8;
    afr[ks] = cat8(*(const f16x8*)p, *(const f16x8*)(p + 16));
  }

  int flr[8];
  #pragma unroll
  for (int r = 0; r < 8; ++r) flr[r] = s_fl[wv * 16 + r + 8 * h];

  // ---- embedding panel must be fully resident before matmul2 ----
#if EAD_ASYNC_STAGE
  asm volatile("s_wait_asynccnt 0" ::: "memory");
#endif
  __syncthreads();

  // ---- matmul2: weights @ emb_table, 48 D-tiles, fused select + store ----
  #pragma unroll 2
  for (int dt = 0; dt < 48; ++dt) {
    f32x8 c = {};
    #pragma unroll
    for (int ks = 0; ks < 4; ++ks) {
      const _Float16* p = &s_emb[(dt * 16 + l16) * KPAD + ks * 32 + h * 8];
      f16x16 bfr = cat8(*(const f16x8*)p, *(const f16x8*)(p + 16));
      c = __builtin_amdgcn_wmma_f32_16x16x32_f16(false, afr[ks], false, bfr,
                                                 (short)0, c, false, false);
    }
    const int d = dt * 16 + l16;
    const float pv = s_pad[d];
    const float mv = s_mask[d];
    #pragma unroll
    for (int r = 0; r < 8; ++r) {
      float val = c[r];
      int f = flr[r];
      if (f & 1) val = pv;     // pad first
      if (f & 2) val = mv;     // mask wins (applied second in reference)
      out[(size_t)(tokBase + r + 8 * h) * DIM + d] = val;
    }
  }
}

// ---------------------------------------------------------------------------
extern "C" void kernel_launch(void* const* d_in, const int* in_sizes, int n_in,
                              void* d_out, int out_size, void* d_ws, size_t ws_size,
                              hipStream_t stream) {
  const float* gene  = (const float*)d_in[0];
  const int*   padm  = (const int*)  d_in[1];
  const int*   mskm  = (const int*)  d_in[2];
  const float* w1    = (const float*)d_in[3];
  const float* b1    = (const float*)d_in[4];
  const float* w2    = (const float*)d_in[5];
  const float* b2    = (const float*)d_in[6];
  const float* embt  = (const float*)d_in[7];
  const float* pade  = (const float*)d_in[8];
  const float* maske = (const float*)d_in[9];
  float* out = (float*)d_out;

  _Float16* w2h   = (_Float16*)d_ws;                       // 112*128*2 = 28 KB
  _Float16* embTh = (_Float16*)((char*)d_ws + 32768);      // 768*128*2 = 192 KB

  ead_prep<<<(DIM * KPAD + 255) / 256, 256, 0, stream>>>(w2, embt, w2h, embTh);
  ead_main<<<NBLOCKS, 256, 0, stream>>>(gene, padm, mskm, w1, b1, b2,
                                        pade, maske, w2h, embTh, out);
}
